// GraphConvolution_44753559224578
// MI455X (gfx1250) — compile-verified, matrix operand order checked by reading, not run on hardware
//
#include <hip/hip_runtime.h>
#include <hip/hip_bf16.h>

// GCN layer for MI455X (gfx1250, wave32):
//   out = (x @ W1^T + b1) + support @ (x @ W2^T)
// Strategy: bf16 V_WMMA_F32_16X16X32_BF16 with fp32 accumulate.
// fp32 -> bf16 conversion happens during LDS staging (no extra HBM pass).
// Stage 1 fuses both x-GEMMs (x read once); x2 intermediate kept as bf16
// in workspace (~15.7 MB, L2-resident for stage 2).

typedef __bf16 bf16_t;
typedef __attribute__((ext_vector_type(16))) __bf16 v16bf;
typedef __attribute__((ext_vector_type(8)))  float  v8f;

#define B_SZ     16
#define N_SZ     2562
#define DIN      963
#define DOUT     192
#define M_TOTAL  (B_SZ * N_SZ)      // 40992

#define TILE_M   64
#define TILE_K   32
#define LDSS     34                 // shorts per LDS row (17 dwords, conflict-free)
#define NT       12                 // 192 / 16 n-tiles

static __device__ __forceinline__ unsigned short f32_to_bf16_rne(float f) {
    union { float f; unsigned int u; } v; v.f = f;
    unsigned int u = v.u;
    return (unsigned short)((u + 0x7FFFu + ((u >> 16) & 1u)) >> 16);
}

union FragU { v16bf v; unsigned int u[8]; };

// Load one 16-lane-group operand fragment from an LDS tile laid out [rows][LDSS].
// ISA 7.12.2 (16-bit 16x32 A layout, wave32): lane L holds row (L&15);
// half = L>>4 selects K-subgroup; vector pair p covers k = 2p + 8*half (+8 for p>=4).
// B operand mirrors this with "row" = output column (B^T row).
static __device__ __forceinline__ v16bf load_frag(const unsigned short* lds, int row, int half) {
    const unsigned int* p = (const unsigned int*)(lds + row * LDSS);  // 68B row stride: dword aligned
    FragU fr;
#pragma unroll
    for (int q = 0; q < 8; ++q) {
        int kp = 2 * q + 8 * half + ((q >= 4) ? 8 : 0);   // even k of the pair
        fr.u[q] = p[kp >> 1];
    }
    return fr.v;
}

// ---------------------------------------------------------------------------
// Stage 1: y1 = x@W1^T + b1 -> out (fp32);  x2 = x@W2^T -> ws (bf16)
// 256 threads = 8 waves: wave = (m_sub 0..3) x (wsel: 0=W1, 1=W2)
// ---------------------------------------------------------------------------
__global__ __launch_bounds__(256)
void gcn_lin_dual(const float* __restrict__ x,      // [M_TOTAL][DIN]
                  const float* __restrict__ W1,     // [DOUT][DIN]
                  const float* __restrict__ b1,     // [DOUT]
                  const float* __restrict__ W2,     // [DOUT][DIN]
                  float* __restrict__ out,          // [M_TOTAL][DOUT]
                  unsigned short* __restrict__ x2b) // [M_TOTAL][DOUT] bf16
{
    __shared__ unsigned short sA[TILE_M * LDSS];        // x tile (bf16)
    __shared__ unsigned short sW[2 * DOUT * LDSS];      // W1 & W2 tiles (bf16)

    const int tid   = threadIdx.x;
    const int lane  = tid & 31;
    const int wave  = tid >> 5;
    const int m_sub = wave & 3;
    const int wsel  = wave >> 2;
    const int m0    = blockIdx.x * TILE_M;
    const int row16 = lane & 15;
    const int half  = lane >> 4;

    v8f acc[NT];
#pragma unroll
    for (int t = 0; t < NT; ++t) acc[t] = (v8f)0.0f;

    for (int k0 = 0; k0 < DIN; k0 += TILE_K) {
        // stage x tile: TILE_M x TILE_K, fp32 -> bf16, zero-fill edges
#pragma unroll
        for (int e = tid; e < TILE_M * TILE_K; e += 256) {
            int r = e >> 5, c = e & 31;
            int m = m0 + r, k = k0 + c;
            float v = (m < M_TOTAL && k < DIN) ? x[(long)m * DIN + k] : 0.0f;
            sA[r * LDSS + c] = f32_to_bf16_rne(v);
        }
        // stage both weight tiles: 2 x DOUT x TILE_K
        for (int e = tid; e < 2 * DOUT * TILE_K; e += 256) {
            int w   = e / (DOUT * TILE_K);
            int rem = e - w * (DOUT * TILE_K);
            int f = rem >> 5, c = rem & 31;
            int k = k0 + c;
            const float* Wp = (w == 0) ? W1 : W2;
            float v = (k < DIN) ? Wp[(long)f * DIN + k] : 0.0f;
            sW[(w * DOUT + f) * LDSS + c] = f32_to_bf16_rne(v);
        }
        // prefetch next x tile (global_prefetch_b8; speculative, safe)
        if (k0 + TILE_K < DIN) {
            int pm = m0 + (tid & 63);
            if (pm < M_TOTAL) __builtin_prefetch(&x[(long)pm * DIN + k0 + TILE_K], 0, 0);
        }
        __syncthreads();

        v16bf a = load_frag(sA, 16 * m_sub + row16, half);
        const unsigned short* wbase = sW + wsel * DOUT * LDSS;
#pragma unroll
        for (int t = 0; t < NT; ++t) {
            v16bf b = load_frag(wbase, 16 * t + row16, half);
            acc[t] = __builtin_amdgcn_wmma_f32_16x16x32_bf16(
                false, a, false, b, (short)0, acc[t], false, false);
        }
        __syncthreads();
    }

    // C/D layout: lane holds column n=row16; VGPR r -> row m = r + 8*half
#pragma unroll
    for (int t = 0; t < NT; ++t) {
        int f = 16 * t + row16;
        float bias = (wsel == 0) ? b1[f] : 0.0f;
#pragma unroll
        for (int r = 0; r < 8; ++r) {
            int m = m0 + 16 * m_sub + 8 * half + r;
            if (m < M_TOTAL) {
                if (wsel == 0) out[(long)m * DOUT + f] = acc[t][r] + bias;
                else           x2b[(long)m * DOUT + f] = f32_to_bf16_rne(acc[t][r]);
            }
        }
    }
}

// ---------------------------------------------------------------------------
// Stage 2: out[b] += support @ x2[b]   (per-batch 2562x2562x192 GEMM)
// 8 waves: (m_sub 0..3) x (tgrp 0..1, each owning 6 of the 12 n-tiles)
// ---------------------------------------------------------------------------
#define NT3 6

__global__ __launch_bounds__(256)
void gcn_aggregate(const float* __restrict__ support,       // [N_SZ][N_SZ]
                   const unsigned short* __restrict__ x2b,  // [B][N_SZ][DOUT] bf16
                   float* __restrict__ out)                 // [B][N_SZ][DOUT]
{
    __shared__ unsigned short sA[TILE_M * LDSS];   // support tile (bf16)
    __shared__ unsigned short sB[DOUT * LDSS];     // x2 tile, transposed [f][k]

    const int tid   = threadIdx.x;
    const int lane  = tid & 31;
    const int wave  = tid >> 5;
    const int m_sub = wave & 3;
    const int tgrp  = wave >> 2;
    const int m0    = blockIdx.x * TILE_M;
    const int b     = blockIdx.y;
    const int row16 = lane & 15;
    const int half  = lane >> 4;

    v8f acc[NT3];
#pragma unroll
    for (int t = 0; t < NT3; ++t) acc[t] = (v8f)0.0f;

    const unsigned short* x2base = x2b + (long)b * N_SZ * DOUT;

    for (int k0 = 0; k0 < N_SZ; k0 += TILE_K) {
        // support tile: fp32 -> bf16, zero-fill edges
#pragma unroll
        for (int e = tid; e < TILE_M * TILE_K; e += 256) {
            int r = e >> 5, c = e & 31;
            int m = m0 + r, k = k0 + c;
            float v = (m < N_SZ && k < N_SZ) ? support[(long)m * N_SZ + k] : 0.0f;
            sA[r * LDSS + c] = f32_to_bf16_rne(v);
        }
        // x2 tile [TILE_K][DOUT] bf16 -> LDS transposed [DOUT][TILE_K]
        // consecutive tid -> consecutive f: coalesced global reads
        for (int e = tid; e < TILE_K * DOUT; e += 256) {
            int kc = e / DOUT;
            int f  = e - kc * DOUT;
            int k  = k0 + kc;
            unsigned short v = (k < N_SZ) ? x2base[(long)k * DOUT + f] : (unsigned short)0;
            sB[f * LDSS + kc] = v;
        }
        if (k0 + TILE_K < N_SZ) {
            int pm = m0 + (tid & 63);
            if (pm < N_SZ) __builtin_prefetch(&support[(long)pm * N_SZ + k0 + TILE_K], 0, 0);
        }
        __syncthreads();

        v16bf a = load_frag(sA, 16 * m_sub + row16, half);
#pragma unroll
        for (int t = 0; t < NT3; ++t) {
            v16bf bb = load_frag(sB, 16 * (tgrp * NT3 + t) + row16, half);
            acc[t] = __builtin_amdgcn_wmma_f32_16x16x32_bf16(
                false, a, false, bb, (short)0, acc[t], false, false);
        }
        __syncthreads();
    }

    // accumulate into out (each (b,m,f) owned by exactly one wave; stage 1
    // rewrites out every call, so the RMW is deterministic across replays)
#pragma unroll
    for (int t = 0; t < NT3; ++t) {
        int f = 16 * (tgrp * NT3 + t) + row16;
#pragma unroll
        for (int r = 0; r < 8; ++r) {
            int m = m0 + 16 * m_sub + 8 * half + r;
            if (m < N_SZ) {
                long o = ((long)b * N_SZ + m) * DOUT + f;
                out[o] += acc[t][r];
            }
        }
    }
}

extern "C" void kernel_launch(void* const* d_in, const int* in_sizes, int n_in,
                              void* d_out, int out_size, void* d_ws, size_t ws_size,
                              hipStream_t stream) {
    (void)in_sizes; (void)n_in; (void)out_size; (void)ws_size;
    const float* x       = (const float*)d_in[0];
    const float* support = (const float*)d_in[1];
    const float* W1      = (const float*)d_in[2];
    const float* b1      = (const float*)d_in[3];
    const float* W2      = (const float*)d_in[4];
    float* out           = (float*)d_out;
    // workspace: bf16 x2 = [B][N][DOUT] -> 40992*192*2 = 15,740,928 bytes
    unsigned short* x2b  = (unsigned short*)d_ws;

    dim3 g1((M_TOTAL + TILE_M - 1) / TILE_M);           // 641
    gcn_lin_dual<<<g1, 256, 0, stream>>>(x, W1, b1, W2, out, x2b);

    dim3 g2((N_SZ + TILE_M - 1) / TILE_M, B_SZ);        // 41 x 16
    gcn_aggregate<<<g2, 256, 0, stream>>>(support, x2b, out);
}
